// EGCN_45853070852215
// MI455X (gfx1250) — matrix operand order repaired and checked by reading, not compiled, not run on hardware
//
#include <hip/hip_runtime.h>
#include <hip/hip_bf16.h>
#include <math.h>

#define NNODES 50000
#define NEDGES 800000
#define HDF    256     // H*D
#define DOUT   64
#define KIN    64
#define NBLK   3
#define EPSV   1e-5f

typedef __attribute__((ext_vector_type(16))) _Float16 v16h;
typedef __attribute__((ext_vector_type(8)))  _Float16 v8h;
typedef __attribute__((ext_vector_type(8)))  float    v8f;

__device__ __forceinline__ float gelu_exact(float x) {
    return 0.5f * x * (1.0f + erff(x * 0.70710678118654752f));
}

// A-fragment (16x32, f16) from an LDS row-major tile with `stride` halves.
// A layout (contiguous, matches packed B): lanes 0-15: K = kbase+{0..15}; lanes 16-31: +16.
__device__ __forceinline__ v16h load_a_frag(const _Float16* As, int stride, int kbase, int lane) {
    int row = lane & 15;
    int hi  = (lane >> 4) << 4;            // 0 or 16
    const _Float16* p = As + row * stride + kbase + hi;
    v8h lo = *(const v8h*)(p);
    v8h hs = *(const v8h*)(p + 8);
    v16h a;
#pragma unroll
    for (int i = 0; i < 8; ++i) { a[i] = lo[i]; a[i + 8] = hs[i]; }
    return a;
}

// Pack fp32 weights W[NBLK][Ktot][Ncols] into per-lane-contiguous f16 B-fragments.
// Fragment (b, nt, ks): P[frag*512 + lane*16 + h] = W[b][ks*32 + (lane>=16?16:0) + h][nt*16 + lane%16]
__global__ void pack_weights(const float* __restrict__ W, _Float16* __restrict__ P,
                             int Ktot, int Ncols, int ntiles, int ksteps) {
    int idx = blockIdx.x * blockDim.x + threadIdx.x;
    int total = NBLK * ntiles * ksteps * 512;
    if (idx >= total) return;
    int within = idx & 511;
    int frag   = idx >> 9;
    int lane = within >> 4;
    int h    = within & 15;
    int ks   = frag % ksteps;
    int nt   = (frag / ksteps) % ntiles;
    int b    = frag / (ksteps * ntiles);
    int n = nt * 16 + (lane & 15);
    int k = ks * 32 + ((lane >> 4) << 4) + h;
    P[idx] = (_Float16)W[((long)b * Ktot + k) * Ncols + n];
}

__global__ void zero_f4(float4* __restrict__ p, int n4) {
    int i = blockIdx.x * blockDim.x + threadIdx.x;
    if (i < n4) p[i] = make_float4(0.f, 0.f, 0.f, 0.f);
}

// hv = LN(GELU((x0 [+ belta*x0 resid scheme]) @ Wn)) ; 16 rows per workgroup, 8 waves x 32 cols.
__global__ __launch_bounds__(256) void node_proj(
    const float* __restrict__ x0, const float* __restrict__ out0,
    const float* __restrict__ beltas, int bidx,
    const _Float16* __restrict__ wpack,
    const float* __restrict__ lng, const float* __restrict__ lnb,
    float* __restrict__ hv)
{
    __shared__ alignas(16) _Float16 As[16 * 72];
    __shared__ alignas(16) float obuf[16 * (HDF + 8)];
    const int OBS = HDF + 8;
    int tid = threadIdx.x;
    int rowbase = blockIdx.x * 16;
    float belta = (out0 != nullptr) ? beltas[bidx] : 0.f;

    for (int t = tid; t < 16 * KIN; t += 256) {
        int r = t >> 6, c = t & 63;
        long gi = (long)(rowbase + r) * KIN + c;
        float v = x0[gi];
        if (out0) v = out0[gi] + belta * v;       // block input = out0 + belta * node_feats
        As[r * 72 + c] = (_Float16)v;
    }
    __syncthreads();

    int wave = tid >> 5, lane = tid & 31;
#pragma unroll
    for (int tt = 0; tt < 2; ++tt) {
        int nt = wave * 2 + tt;
        v8f acc = {};
#pragma unroll
        for (int ks = 0; ks < 2; ++ks) {
            v16h a  = load_a_frag(As, 72, ks * 32, lane);
            v16h bm = *(const v16h*)(wpack + (nt * 2 + ks) * 512 + lane * 16);
            acc = __builtin_amdgcn_wmma_f32_16x16x32_f16(false, a, false, bm,
                                                         (short)0, acc, false, false);
        }
        int col = nt * 16 + (lane & 15);
        int mb  = (lane >> 4) << 3;
#pragma unroll
        for (int i = 0; i < 8; ++i)
            obuf[(mb + i) * OBS + col] = gelu_exact(acc[i]);
    }
    __syncthreads();

    // LN over 256 per row: 16 lanes per row
    int row = tid >> 4, s = tid & 15;
    float sum = 0.f, sq = 0.f;
    for (int c = s; c < HDF; c += 16) { float v = obuf[row * OBS + c]; sum += v; sq += v * v; }
#pragma unroll
    for (int off = 1; off < 16; off <<= 1) {
        sum += __shfl_xor(sum, off, 16);
        sq  += __shfl_xor(sq,  off, 16);
    }
    float mean = sum * (1.f / HDF);
    float var  = sq * (1.f / HDF) - mean * mean;
    float rstd = rsqrtf(var + EPSV);
    float* orow = hv + (long)(rowbase + row) * HDF;
    for (int c = s; c < HDF; c += 16)
        orow[c] = (obuf[row * OBS + c] - mean) * rstd * lng[c] + lnb[c];
}

// Fused: he = exp(LN(e @ We)); h[dst] += hv[src] * he   (16 edges per workgroup)
__global__ __launch_bounds__(256) void edge_fused(
    const float* __restrict__ ef, const int* __restrict__ src, const int* __restrict__ dst,
    const _Float16* __restrict__ wpack,
    const float* __restrict__ lng, const float* __restrict__ lnb,
    const float* __restrict__ hv, float* __restrict__ h)
{
    __shared__ alignas(16) _Float16 As[16 * 72];
    __shared__ alignas(16) float obuf[16 * (HDF + 8)];
    __shared__ int sidx[16], didx[16];
    const int OBS = HDF + 8;
    int tid = threadIdx.x;
    int ebase = blockIdx.x * 16;
    if (tid < 16) { sidx[tid] = src[ebase + tid]; didx[tid] = dst[ebase + tid]; }
    for (int t = tid; t < 16 * KIN; t += 256) {
        int r = t >> 6, c = t & 63;
        As[r * 72 + c] = (_Float16)ef[(long)(ebase + r) * KIN + c];
    }
    __syncthreads();

    int wave = tid >> 5, lane = tid & 31;
#pragma unroll
    for (int tt = 0; tt < 2; ++tt) {
        int nt = wave * 2 + tt;
        v8f acc = {};
#pragma unroll
        for (int ks = 0; ks < 2; ++ks) {
            v16h a  = load_a_frag(As, 72, ks * 32, lane);
            v16h bm = *(const v16h*)(wpack + (nt * 2 + ks) * 512 + lane * 16);
            acc = __builtin_amdgcn_wmma_f32_16x16x32_f16(false, a, false, bm,
                                                         (short)0, acc, false, false);
        }
        int col = nt * 16 + (lane & 15);
        int mb  = (lane >> 4) << 3;
#pragma unroll
        for (int i = 0; i < 8; ++i)
            obuf[(mb + i) * OBS + col] = acc[i];          // no GELU on edge path
    }
    __syncthreads();

    int row = tid >> 4, s = tid & 15;
    float sum = 0.f, sq = 0.f;
    for (int c = s; c < HDF; c += 16) { float v = obuf[row * OBS + c]; sum += v; sq += v * v; }
#pragma unroll
    for (int off = 1; off < 16; off <<= 1) {
        sum += __shfl_xor(sum, off, 16);
        sq  += __shfl_xor(sq,  off, 16);
    }
    float mean = sum * (1.f / HDF);
    float var  = sq * (1.f / HDF) - mean * mean;
    float rstd = rsqrtf(var + EPSV);
    for (int c = s; c < HDF; c += 16) {
        float v = (obuf[row * OBS + c] - mean) * rstd * lng[c] + lnb[c];
        obuf[row * OBS + c] = expf(v);                    // he kept in LDS only
    }
    __syncthreads();

    // gather hv[src], multiply, atomic scatter-add to h[dst]; float4 chunks
    for (int t = tid; t < 16 * 64; t += 256) {
        int el = t >> 6;
        int c4 = (t & 63) << 2;
        int sv = sidx[el], dv = didx[el];
        const float4 hvv = *(const float4*)(hv + (long)sv * HDF + c4);
        const float4 hev = *(const float4*)(obuf + el * OBS + c4);
        float* hp = h + (long)dv * HDF + c4;
        atomicAdd(hp + 0, hvv.x * hev.x);
        atomicAdd(hp + 1, hvv.y * hev.y);
        atomicAdd(hp + 2, hvv.z * hev.z);
        atomicAdd(hp + 3, hvv.w * hev.w);
    }
}

// y = LN(GELU(h @ Wo)); out = (b==0) ? y : out + alpha*y   (out doubles as out0)
__global__ __launch_bounds__(256) void out_proj(
    const float* __restrict__ h,
    const _Float16* __restrict__ wpack,
    const float* __restrict__ lng, const float* __restrict__ lnb,
    const float* __restrict__ alphas, int bidx,
    float* __restrict__ out)
{
    __shared__ alignas(16) _Float16 As[16 * 264];
    __shared__ alignas(16) float obuf[16 * (DOUT + 8)];
    const int OBS = DOUT + 8;
    int tid = threadIdx.x;
    int rowbase = blockIdx.x * 16;

    for (int t = tid; t < 16 * HDF; t += 256) {
        int r = t >> 8, c = t & 255;
        As[r * 264 + c] = (_Float16)h[(long)(rowbase + r) * HDF + c];
    }
    __syncthreads();

    int wave = tid >> 5, lane = tid & 31;
    if (wave < 4) {
        int nt = wave;
        v8f acc = {};
#pragma unroll
        for (int ks = 0; ks < 8; ++ks) {
            v16h a  = load_a_frag(As, 264, ks * 32, lane);
            v16h bm = *(const v16h*)(wpack + (nt * 8 + ks) * 512 + lane * 16);
            acc = __builtin_amdgcn_wmma_f32_16x16x32_f16(false, a, false, bm,
                                                         (short)0, acc, false, false);
        }
        int col = nt * 16 + (lane & 15);
        int mb  = (lane >> 4) << 3;
#pragma unroll
        for (int i = 0; i < 8; ++i)
            obuf[(mb + i) * OBS + col] = gelu_exact(acc[i]);
    }
    __syncthreads();

    int row = tid >> 4, s = tid & 15;
    float sum = 0.f, sq = 0.f;
    for (int c = s; c < DOUT; c += 16) { float v = obuf[row * OBS + c]; sum += v; sq += v * v; }
#pragma unroll
    for (int off = 1; off < 16; off <<= 1) {
        sum += __shfl_xor(sum, off, 16);
        sq  += __shfl_xor(sq,  off, 16);
    }
    float mean = sum * (1.f / DOUT);
    float var  = sq * (1.f / DOUT) - mean * mean;
    float rstd = rsqrtf(var + EPSV);
    float alpha = (bidx > 0) ? alphas[bidx] : 0.f;
    float* orow = out + (long)(rowbase + row) * DOUT;
    for (int c = s; c < DOUT; c += 16) {
        float y = (obuf[row * OBS + c] - mean) * rstd * lng[c] + lnb[c];
        orow[c] = (bidx > 0) ? (orow[c] + alpha * y) : y;
    }
}

extern "C" void kernel_launch(void* const* d_in, const int* in_sizes, int n_in,
                              void* d_out, int out_size, void* d_ws, size_t ws_size,
                              hipStream_t stream) {
    const float* node_feats = (const float*)d_in[0];
    const float* edge_feats = (const float*)d_in[1];
    const int*   src        = (const int*)d_in[2];
    const int*   dst        = (const int*)d_in[3];
    const float* Wn         = (const float*)d_in[4];
    const float* We         = (const float*)d_in[5];
    const float* Wo         = (const float*)d_in[6];
    const float* ln_n_g     = (const float*)d_in[7];
    const float* ln_n_b     = (const float*)d_in[8];
    const float* ln_e_g     = (const float*)d_in[9];
    const float* ln_e_b     = (const float*)d_in[10];
    const float* ln_o_g     = (const float*)d_in[11];
    const float* ln_o_b     = (const float*)d_in[12];
    const float* alphas     = (const float*)d_in[13];
    const float* beltas     = (const float*)d_in[14];
    float* out = (float*)d_out;

    char* ws = (char*)d_ws;
    size_t off = 0;
    auto alloc = [&](size_t bytes) -> void* {
        void* p = ws + off;
        off = (off + bytes + 255) & ~(size_t)255;
        return p;
    };
    _Float16* packWn = (_Float16*)alloc((size_t)NBLK * 32 * 512 * 2);
    _Float16* packWe = (_Float16*)alloc((size_t)NBLK * 32 * 512 * 2);
    _Float16* packWo = (_Float16*)alloc((size_t)NBLK * 32 * 512 * 2);
    float* hv = (float*)alloc((size_t)NNODES * HDF * 4);
    float* h  = (float*)alloc((size_t)NNODES * HDF * 4);

    int packN = NBLK * 32 * 512;
    pack_weights<<<(packN + 255) / 256, 256, 0, stream>>>(Wn, packWn, KIN, HDF, 16, 2);
    pack_weights<<<(packN + 255) / 256, 256, 0, stream>>>(We, packWe, KIN, HDF, 16, 2);
    pack_weights<<<(packN + 255) / 256, 256, 0, stream>>>(Wo, packWo, HDF, DOUT, 4, 8);

    for (int b = 0; b < NBLK; ++b) {
        node_proj<<<NNODES / 16, 256, 0, stream>>>(
            node_feats, b ? out : nullptr, beltas, b,
            packWn + (size_t)b * 32 * 512, ln_n_g + b * HDF, ln_n_b + b * HDF, hv);
        int n4 = NNODES * HDF / 4;
        zero_f4<<<(n4 + 255) / 256, 256, 0, stream>>>((float4*)h, n4);
        edge_fused<<<NEDGES / 16, 256, 0, stream>>>(
            edge_feats, src, dst,
            packWe + (size_t)b * 32 * 512, ln_e_g + b * HDF, ln_e_b + b * HDF, hv, h);
        out_proj<<<NNODES / 16, 256, 0, stream>>>(
            h, packWo + (size_t)b * 32 * 512, ln_o_g + b * DOUT, ln_o_b + b * DOUT,
            alphas, b, out);
    }
}
